// RST_63634235458231
// MI455X (gfx1250) — matrix operand order repaired and checked by reading, not compiled, not run on hardware
//
#include <hip/hip_runtime.h>
#include <hip/hip_bf16.h>

// Tree-GRU reduction for MI455X (gfx1250, wave32, WMMA).
// Per level: one fused kernel does 4 GEMMs ([16x256]x[256x768] per block) with
// v_wmma_f32_16x16x32_bf16 + both GRU gate stages, keeping gi/gh/h1 on chip.

#define D 256
#define GD 768           // 3*D
#define TM 16            // pair-rows per workgroup
#define NTHREADS 128     // 4 wave32s

typedef __attribute__((ext_vector_type(16))) __bf16 v16bf;
typedef __attribute__((ext_vector_type(8)))  float  v8f;

union Frag {
    v16bf bf;
    unsigned u[8];
    uint4 q[2];
};

// round-to-nearest-even f32 -> bf16, packed pair (x -> [15:0], y -> [31:16])
__device__ __forceinline__ unsigned pack_bf16(float x, float y) {
    unsigned a = __float_as_uint(x);
    unsigned b = __float_as_uint(y);
    a = (a + 0x7fffu + ((a >> 16) & 1u)) >> 16;
    b = (b + 0x7fffu + ((b >> 16) & 1u)) & 0xffff0000u;
    return (a & 0xffffu) | b;
}

__device__ __forceinline__ unsigned short cvt_bf16(float x) {
    unsigned a = __float_as_uint(x);
    return (unsigned short)((a + 0x7fffu + ((a >> 16) & 1u)) >> 16);
}

// A fragment: 16x32 bf16, M = lane&15, each lane half holds 16 contiguous K (k = 16*half + s).
// row: f32 row of length 256 (global or LDS), k0: K tile base.
__device__ __forceinline__ void loadA(Frag& f, const float* row, int k0, int half) {
    const float4* p0 = reinterpret_cast<const float4*>(row + k0 + half * 16);
    const float4* p1 = reinterpret_cast<const float4*>(row + k0 + half * 16 + 8);
    float4 f0 = p0[0], f1 = p0[1];
    float4 f2 = p1[0], f3 = p1[1];
    f.u[0] = pack_bf16(f0.x, f0.y); f.u[1] = pack_bf16(f0.z, f0.w);
    f.u[2] = pack_bf16(f1.x, f1.y); f.u[3] = pack_bf16(f1.z, f1.w);
    f.u[4] = pack_bf16(f2.x, f2.y); f.u[5] = pack_bf16(f2.z, f2.w);
    f.u[6] = pack_bf16(f3.x, f3.y); f.u[7] = pack_bf16(f3.z, f3.w);
}

// B fragment: 32x16 bf16 (K x N). W is bf16 [768 x 256] row-major, i.e. W[n][k]
// = B[k][n]; each lane (N = lane&15) reads 16 contiguous K values (32 bytes).
__device__ __forceinline__ void loadB(Frag& f, const unsigned short* W, int n, int k0, int half) {
    const uint4* p = reinterpret_cast<const uint4*>(W + (size_t)n * D + k0 + half * 16);
    f.q[0] = p[0];
    f.q[1] = p[1];
}

__device__ __forceinline__ v8f wmma_bf16(const Frag& a, const Frag& b, v8f c) {
    return __builtin_amdgcn_wmma_f32_16x16x32_bf16(
        false, a.bf, false, b.bf, (short)0, c, false, false);
}

__device__ __forceinline__ float sigmoidf_(float x) {
    return 1.0f / (1.0f + __expf(-x));
}

__global__ __launch_bounds__(NTHREADS)
void tree_gru_level(const float* __restrict__ embeds_in,
                    const float* __restrict__ hs_in,   // nullptr -> zeros (level 0)
                    float* __restrict__ embeds_out,
                    float* __restrict__ hs_out,
                    const unsigned short* __restrict__ Wih,  // bf16 [768x256]
                    const unsigned short* __restrict__ Whh,  // bf16 [768x256]
                    const float* __restrict__ b_ih,
                    const float* __restrict__ b_hh,
                    int half_n) {
    __shared__ __align__(16) float sH0[TM][D];
    __shared__ __align__(16) float sH1[TM][D];

    const int tid  = threadIdx.x;
    const int wave = tid >> 5;
    const int lane = tid & 31;
    const int half = lane >> 4;
    const int nl   = lane & 15;
    const int n_in = half_n * 2;

    // per-lane pair-row pointers (row index = lane&15)
    {
        int p = blockIdx.x * TM + nl;
        int b = p / half_n;
        int j = p - b * half_n;
        (void)b; (void)j;
    }
    const int p_row = blockIdx.x * TM + nl;
    const int b_row = p_row / half_n;
    const int j_row = p_row - b_row * half_n;
    const float* xLrow = embeds_in + ((size_t)b_row * n_in + 2 * j_row) * D;
    const float* xRrow = xLrow + D;

    // cooperative: sH0 = 0.5*(hL + hR)  (zeros at level 0)
    for (int e = tid; e < TM * D; e += NTHREADS) {
        int mm  = e >> 8;
        int col = e & (D - 1);
        int p   = blockIdx.x * TM + mm;
        int b   = p / half_n;
        int j   = p - b * half_n;
        float v = 0.0f;
        if (hs_in != nullptr) {
            const float* hl = hs_in + ((size_t)b * n_in + 2 * j) * D;
            v = 0.5f * (hl[col] + hl[D + col]);
        }
        sH0[mm][col] = v;
    }
    __syncthreads();

    // ---------------- stage 1: h1 = GRU(xL, h0) ----------------
    #pragma unroll 1
    for (int ct = 0; ct < 4; ++ct) {
        const int c0 = wave * 64 + ct * 16;   // column base within a gate section
        v8f a_ir = {0,0,0,0,0,0,0,0}, a_iz = a_ir, a_in = a_ir;
        v8f a_hr = a_ir, a_hz = a_ir, a_hn = a_ir;
        #pragma unroll
        for (int k0 = 0; k0 < D; k0 += 32) {
            Frag ax, ah, br, bz, bn, cr, cz, cn;
            loadA(ax, xLrow, k0, half);
            loadA(ah, &sH0[nl][0], k0, half);
            loadB(br, Wih, c0,           k0, half);
            loadB(bz, Wih, c0 + D,       k0, half);
            loadB(bn, Wih, c0 + 2 * D,   k0, half);
            loadB(cr, Whh, c0,           k0, half);
            loadB(cz, Whh, c0 + D,       k0, half);
            loadB(cn, Whh, c0 + 2 * D,   k0, half);
            a_ir = wmma_bf16(ax, br, a_ir);
            a_iz = wmma_bf16(ax, bz, a_iz);
            a_in = wmma_bf16(ax, bn, a_in);
            a_hr = wmma_bf16(ah, cr, a_hr);
            a_hz = wmma_bf16(ah, cz, a_hz);
            a_hn = wmma_bf16(ah, cn, a_hn);
        }
        const int col = c0 + nl;
        const float bir = b_ih[col], biz = b_ih[col + D], bin = b_ih[col + 2 * D];
        const float bhr = b_hh[col], bhz = b_hh[col + D], bhn = b_hh[col + 2 * D];
        #pragma unroll
        for (int r = 0; r < 8; ++r) {
            const int mm = r + 8 * half;          // C layout: M = vgpr + 8*half, N = lane&15
            float ir = a_ir[r] + bir, iz = a_iz[r] + biz, inn = a_in[r] + bin;
            float hr = a_hr[r] + bhr, hz = a_hz[r] + bhz, hn  = a_hn[r] + bhn;
            float rg = sigmoidf_(ir + hr);
            float zg = sigmoidf_(iz + hz);
            float ng = tanhf(inn + rg * hn);
            float h0v = sH0[mm][col];
            sH1[mm][col] = (1.0f - zg) * ng + zg * h0v;
        }
    }
    __syncthreads();

    // ---------------- stage 2: h2 = GRU(xR, h1) ----------------
    #pragma unroll 1
    for (int ct = 0; ct < 4; ++ct) {
        const int c0 = wave * 64 + ct * 16;
        v8f a_ir = {0,0,0,0,0,0,0,0}, a_iz = a_ir, a_in = a_ir;
        v8f a_hr = a_ir, a_hz = a_ir, a_hn = a_ir;
        #pragma unroll
        for (int k0 = 0; k0 < D; k0 += 32) {
            Frag ax, ah, br, bz, bn, cr, cz, cn;
            loadA(ax, xRrow, k0, half);
            loadA(ah, &sH1[nl][0], k0, half);
            loadB(br, Wih, c0,           k0, half);
            loadB(bz, Wih, c0 + D,       k0, half);
            loadB(bn, Wih, c0 + 2 * D,   k0, half);
            loadB(cr, Whh, c0,           k0, half);
            loadB(cz, Whh, c0 + D,       k0, half);
            loadB(cn, Whh, c0 + 2 * D,   k0, half);
            a_ir = wmma_bf16(ax, br, a_ir);
            a_iz = wmma_bf16(ax, bz, a_iz);
            a_in = wmma_bf16(ax, bn, a_in);
            a_hr = wmma_bf16(ah, cr, a_hr);
            a_hz = wmma_bf16(ah, cz, a_hz);
            a_hn = wmma_bf16(ah, cn, a_hn);
        }
        const int col = c0 + nl;
        const float bir = b_ih[col], biz = b_ih[col + D], bin = b_ih[col + 2 * D];
        const float bhr = b_hh[col], bhz = b_hh[col + D], bhn = b_hh[col + 2 * D];
        #pragma unroll
        for (int r = 0; r < 8; ++r) {
            const int mm = r + 8 * half;
            float ir = a_ir[r] + bir, iz = a_iz[r] + biz, inn = a_in[r] + bin;
            float hr = a_hr[r] + bhr, hz = a_hz[r] + bhz, hn  = a_hn[r] + bhn;
            float rg = sigmoidf_(ir + hr);
            float zg = sigmoidf_(iz + hz);
            float ng = tanhf(inn + rg * hn);
            float h1v = sH1[mm][col];
            float h2  = (1.0f - zg) * ng + zg * h1v;
            size_t orow = (size_t)(blockIdx.x * TM + mm) * D;
            embeds_out[orow + col] = 0.5f * (h1v + h2);
            hs_out[orow + col]     = h2;
        }
    }
}

__global__ void f32_to_bf16_kernel(const float* __restrict__ in,
                                   unsigned short* __restrict__ out, int n) {
    int i = blockIdx.x * blockDim.x + threadIdx.x;
    if (i < n) out[i] = cvt_bf16(in[i]);
}

extern "C" void kernel_launch(void* const* d_in, const int* in_sizes, int n_in,
                              void* d_out, int out_size, void* d_ws, size_t ws_size,
                              hipStream_t stream) {
    const float* leaf  = (const float*)d_in[0];   // (32, 4096, 256)
    const float* Wih_f = (const float*)d_in[1];   // (768, 256)
    const float* Whh_f = (const float*)d_in[2];   // (768, 256)
    const float* bih   = (const float*)d_in[3];   // (768,)
    const float* bhh   = (const float*)d_in[4];   // (768,)

    unsigned char* ws = (unsigned char*)d_ws;
    unsigned short* Wih_b = (unsigned short*)(ws);
    unsigned short* Whh_b = (unsigned short*)(ws + (size_t)GD * D * sizeof(unsigned short));

    const size_t DATA_OFF = 1u << 20;             // 1 MB, keeps buffers well aligned
    const size_t ROWS0 = 32ull * 2048;            // largest intermediate (level-0 output)
    const size_t ROWS1 = 32ull * 1024;
    float* e0 = (float*)(ws + DATA_OFF);
    float* h0 = e0 + ROWS0 * D;
    float* e1 = h0 + ROWS0 * D;
    float* h1 = e1 + ROWS1 * D;

    const int wn = GD * D;                        // 196608 per weight matrix
    f32_to_bf16_kernel<<<(wn + 255) / 256, 256, 0, stream>>>(Wih_f, Wih_b, wn);
    f32_to_bf16_kernel<<<(wn + 255) / 256, 256, 0, stream>>>(Whh_f, Whh_b, wn);

    // 12 levels: 4096 -> 1
    for (int l = 0; l < 12; ++l) {
        int n     = 4096 >> l;
        int halfn = n >> 1;
        const float* ein;
        const float* hin;
        if (l == 0)          { ein = leaf; hin = nullptr; }
        else if (l & 1)      { ein = e0;   hin = h0; }    // input = set0 (even-level out)
        else                 { ein = e1;   hin = h1; }    // input = set1 (odd-level out)
        float* eout;
        float* hout;
        if (l & 1) { eout = e1; hout = h1; }
        else       { eout = e0; hout = h0; }
        if (l == 11) eout = (float*)d_out;                // final embeds[:, 0] -> (32, 256)

        int pairs  = 32 * halfn;
        int blocks = pairs / TM;                          // always divisible (pairs >= 32)
        tree_gru_level<<<blocks, NTHREADS, 0, stream>>>(
            ein, hin, eout, hout, Wih_b, Whh_b, bih, bhh, halfn);
    }
}
